// Attention_28604482191565
// MI455X (gfx1250) — matrix operand order repaired and checked by reading, not compiled, not run on hardware
//
#include <hip/hip_runtime.h>
#include <hip/hip_bf16.h>

// ---------------------------------------------------------------------------
// MI455X (gfx1250) transformer block: RMSNorm -> QKV -> qk-norm -> flash attn
// -> O-proj(+res) -> RMSNorm -> GELU MLP(+res).
// Dense math: v_wmma_f32_16x16x32_bf16.
// All GEMM/attention tile staging: double-buffered global_load_async_to_lds_b128
// (ASYNCcnt) so the DMA stream overlaps the WMMA stream.
// ---------------------------------------------------------------------------

#define BB 4
#define SS 2048
#define DD 768
#define HH 12
#define HD 64
#define MLPD 3072
#define BS (BB * SS)          // 8192 rows
#define EPSN 1e-6f

typedef __attribute__((ext_vector_type(16))) __bf16 v16bf;
typedef __attribute__((ext_vector_type(8)))  __bf16 v8bf;
typedef __attribute__((ext_vector_type(8)))  float  v8f;

__device__ __forceinline__ v16bf bconcat(v8bf lo, v8bf hi) {
  return __builtin_shufflevector(lo, hi, 0, 1, 2, 3, 4, 5, 6, 7,
                                 8, 9, 10, 11, 12, 13, 14, 15);
}

// Load a 16-element bf16 fragment as two 16B chunks separated by `gap` elems.
__device__ __forceinline__ v16bf load_frag16(const __bf16* p, int gap) {
  v8bf lo = *reinterpret_cast<const v8bf*>(p);
  v8bf hi = *reinterpret_cast<const v8bf*>(p + gap);
  return bconcat(lo, hi);
}

// Async 16-byte global -> LDS copy (gfx1250, tracked by ASYNCcnt).
// LDS address = low 32 bits of the generic pointer.
__device__ __forceinline__ void async_copy_b128(const __bf16* g, __bf16* l) {
  unsigned lds = (unsigned)(size_t)(void*)l;
  unsigned long long ga = (unsigned long long)(size_t)(const void*)g;
  asm volatile("global_load_async_to_lds_b128 %0, %1, off"
               :: "v"(lds), "v"(ga) : "memory");
}

__device__ __forceinline__ void async_wait0() {
  asm volatile("s_wait_asynccnt 0x0" ::: "memory");
}

__device__ __forceinline__ float gelu_tanh(float x) {
  float x3 = x * x * x;
  return 0.5f * x * (1.0f + tanhf(0.79788456080286535588f * (x + 0.044715f * x3)));
}

// ---------------------------------------------------------------------------
// Tiled transpose + f32->bf16: src[K][N] f32  ->  dst[N][K] bf16.
// One-time per launch for each weight matrix, so GEMM B staging is contiguous.
// ---------------------------------------------------------------------------
__global__ __launch_bounds__(256)
void transpose_f32_to_bf16(const float* __restrict__ src, __bf16* __restrict__ dst,
                           int K, int N) {
  __shared__ float tile[32][33];
  int n0 = blockIdx.x * 32, k0 = blockIdx.y * 32;
  int tx = threadIdx.x & 31, ty = threadIdx.x >> 5;   // 32 x 8
#pragma unroll
  for (int i = 0; i < 32; i += 8)
    tile[ty + i][tx] = src[(size_t)(k0 + ty + i) * N + n0 + tx];
  __syncthreads();
#pragma unroll
  for (int i = 0; i < 32; i += 8)
    dst[(size_t)(n0 + ty + i) * K + k0 + tx] = (__bf16)tile[tx][ty + i];
}

// ---------------------------------------------------------------------------
// Row RMSNorm (f32 in, bf16 out): one block per row.
// ---------------------------------------------------------------------------
__global__ __launch_bounds__(256)
void rmsnorm_to_bf16(const float* __restrict__ x, const float* __restrict__ scale,
                     __bf16* __restrict__ out, int cols) {
  int row = blockIdx.x;
  const float* xr = x + (size_t)row * cols;
  __bf16* orow = out + (size_t)row * cols;

  float ss = 0.f;
  for (int c = threadIdx.x; c < cols; c += blockDim.x) {
    float v = xr[c];
    ss += v * v;
  }
  __shared__ float red[32];
#pragma unroll
  for (int m = 16; m > 0; m >>= 1) ss += __shfl_xor(ss, m, 32);
  int wid = threadIdx.x >> 5, lid = threadIdx.x & 31;
  if (lid == 0) red[wid] = ss;
  __syncthreads();
  if (wid == 0) {
    float v = (lid < (int)(blockDim.x >> 5)) ? red[lid] : 0.f;
#pragma unroll
    for (int m = 16; m > 0; m >>= 1) v += __shfl_xor(v, m, 32);
    if (lid == 0) red[0] = v;
  }
  __syncthreads();
  float rms = rsqrtf(red[0] / (float)cols + EPSN);
  for (int c = threadIdx.x; c < cols; c += blockDim.x)
    orow[c] = (__bf16)(xr[c] * rms * scale[c]);
}

// ---------------------------------------------------------------------------
// qk-norm over head_dim=64, in place on bf16 tensor [BS, H*HD].
// One warp per 64-vector; `extra` folds 1/sqrt(HD) for q.
// ---------------------------------------------------------------------------
__global__ __launch_bounds__(256)
void qk_norm_inplace(__bf16* __restrict__ t, const float* __restrict__ scale,
                     float extra, int nvec) {
  int warp = (blockIdx.x * blockDim.x + threadIdx.x) >> 5;
  int lane = threadIdx.x & 31;
  if (warp >= nvec) return;
  __bf16* v = t + (size_t)warp * HD;
  float a = (float)v[lane];
  float b = (float)v[lane + 32];
  float ss = a * a + b * b;
#pragma unroll
  for (int m = 16; m > 0; m >>= 1) ss += __shfl_xor(ss, m, 32);
  float rms = rsqrtf(ss / (float)HD + EPSN);
  v[lane]      = (__bf16)(a * rms * scale[lane] * extra);
  v[lane + 32] = (__bf16)(b * rms * scale[lane + 32] * extra);
}

// ---------------------------------------------------------------------------
// bf16 WMMA GEMM: C[M,N] = A[M,K] * Bt[N,K]^T   (B pre-transposed)
//   128x128 workgroup tile, 8 waves, wave tile 32x64 (2x4 16x16 frags), KT=32.
//   Double-buffered: 4 async b128 copies per thread per tile overlap WMMA.
// Epilogue (compile-time):
//   1 = f32 + resid | 2 = bf16 | 3 = gelu->bf16 | 4 = bf16, V transposed [B][H][HD][S]
// ---------------------------------------------------------------------------
#define GTM 128
#define GTN 128
#define GTK 32
#define ASTR 40   // padded LDS row stride (elems)
#define BSTR 40

template <int EPI>
__global__ __launch_bounds__(256)
void gemm_bf16_wmma(const __bf16* __restrict__ A, const __bf16* __restrict__ Bt,
                    int M, int N, int K,
                    float* __restrict__ Cf, __bf16* __restrict__ Cb,
                    const float* __restrict__ resid) {
  __shared__ __bf16 Al[2][GTM * ASTR];
  __shared__ __bf16 Bl[2][GTN * BSTR];

  int bn = blockIdx.x * GTN;
  int bm = blockIdx.y * GTM;
  int t = threadIdx.x;
  int wave = t >> 5, lane = t & 31;
  int wr = wave >> 1, wc = wave & 1;       // 4x2 wave grid
  int half = lane >> 4, lr = lane & 15;

  v8f acc[2][4];
#pragma unroll
  for (int mi = 0; mi < 2; mi++)
#pragma unroll
    for (int ni = 0; ni < 4; ni++) acc[mi][ni] = (v8f){};

  // staging: 128 rows x 32 k, 2 threads per row, 16 elems (2 x b128) each
  int srow = t >> 1, scb = (t & 1) * 16;
  const __bf16* Ab = A  + (size_t)(bm + srow) * K + scb;
  const __bf16* Bb = Bt + (size_t)(bn + srow) * K + scb;
  unsigned aoff = srow * ASTR + scb;
  unsigned boff = srow * BSTR + scb;

  int nk = K / GTK;
#define STAGE(buf, k0)                                          \
  do {                                                          \
    async_copy_b128(Ab + (k0),     &Al[buf][aoff]);             \
    async_copy_b128(Ab + (k0) + 8, &Al[buf][aoff + 8]);         \
    async_copy_b128(Bb + (k0),     &Bl[buf][boff]);             \
    async_copy_b128(Bb + (k0) + 8, &Bl[buf][boff + 8]);         \
  } while (0)

  STAGE(0, 0);
  for (int kt = 0; kt < nk; kt++) {
    async_wait0();
    __syncthreads();
    int cur = kt & 1;
    if (kt + 1 < nk) STAGE(cur ^ 1, (kt + 1) * GTK);

    v16bf af[2], bfg[4];
#pragma unroll
    for (int mi = 0; mi < 2; mi++) {
      // A 16x32: lanes 0-15 rows, halves k 0..7 & 16..23 ; lanes 16-31 k+8
      const __bf16* p = &Al[cur][(wr * 32 + mi * 16 + lr) * ASTR + half * 8];
      af[mi] = load_frag16(p, 16);
    }
#pragma unroll
    for (int ni = 0; ni < 4; ni++) {
      // B 32x16: lanes 0-15 col n, k 0..7 & 16..23 ; lanes 16-31 k+8
      const __bf16* p = &Bl[cur][(wc * 64 + ni * 16 + lr) * BSTR + half * 8];
      bfg[ni] = load_frag16(p, 16);
    }
#pragma unroll
    for (int mi = 0; mi < 2; mi++)
#pragma unroll
      for (int ni = 0; ni < 4; ni++)
        acc[mi][ni] = __builtin_amdgcn_wmma_f32_16x16x32_bf16(
            false, af[mi], false, bfg[ni], (short)0, acc[mi][ni], false, false);
  }
#undef STAGE

  // --- epilogue: C frag element (m = half*8 + r, n = lr) ---
#pragma unroll
  for (int mi = 0; mi < 2; mi++)
#pragma unroll
    for (int ni = 0; ni < 4; ni++) {
      int row0 = bm + wr * 32 + mi * 16 + half * 8;
      int col = bn + wc * 64 + ni * 16 + lr;
      if (EPI == 4) {
        // V output transposed to [B][H][HD][S]; 8 consecutive rows = one v8bf
        int bb = row0 >> 11;            // / SS
        int sr = row0 & (SS - 1);
        int hcol = col >> 6;            // / HD
        int hd = col & (HD - 1);
        v8bf pk;
#pragma unroll
        for (int r = 0; r < 8; r++) pk[r] = (__bf16)acc[mi][ni][r];
        *reinterpret_cast<v8bf*>(
            &Cb[((size_t)((bb * HH + hcol) * HD + hd)) * SS + sr]) = pk;
      } else {
#pragma unroll
        for (int r = 0; r < 8; r++) {
          size_t idx = (size_t)(row0 + r) * N + col;
          float v = acc[mi][ni][r];
          if (EPI == 1)      Cf[idx] = v + resid[idx];
          else if (EPI == 2) Cb[idx] = (__bf16)v;
          else               Cb[idx] = (__bf16)gelu_tanh(v);
        }
      }
    }
}

// ---------------------------------------------------------------------------
// Flash attention, non-causal. One wave = 16 query rows; block = 8 waves =
// 128 query rows of one (b,h). Key blocks of 32; K and V^T tiles staged with
// double-buffered async DMA; QK^T and PV on WMMA; online softmax in C-frag
// layout. V is pre-transposed to [B][H][HD][S] by the V-projection epilogue.
// ---------------------------------------------------------------------------
#define KLSTR 72   // K tile LDS row stride (elems), rows = keys
#define VTSTR 40   // V^T tile LDS row stride (elems), rows = head dims
#define PSTR  40   // P patch LDS row stride

__global__ __launch_bounds__(256)
void flash_attn_wmma(const __bf16* __restrict__ Q, const __bf16* __restrict__ Kg,
                     const __bf16* __restrict__ Vg, __bf16* __restrict__ O) {
  __shared__ __bf16 Kl[2][32 * KLSTR];     // [key][hd]
  __shared__ __bf16 Vl[2][HD * VTSTR];     // [hd][key]
  __shared__ __bf16 Pl[8][16 * PSTR];      // per-wave P patch [m][key]

  int b = blockIdx.z, h = blockIdx.y, qt = blockIdx.x;
  int t = threadIdx.x;
  int wave = t >> 5, lane = t & 31;
  int half = lane >> 4, lr = lane & 15;
  int q0 = qt * 128 + wave * 16;

  // Q fragments: rows q0..q0+15, K-dim = HD in two 32-chunks.
  v16bf qf[2];
  {
    size_t qrow = (size_t)(b * SS + q0 + lr) * DD + h * HD;
#pragma unroll
    for (int c = 0; c < 2; c++)
      qf[c] = load_frag16(Q + qrow + c * 32 + half * 8, 16);
  }

  float mrow[8], lrow[8];
  v8f of[4];
#pragma unroll
  for (int r = 0; r < 8; r++) { mrow[r] = -1e30f; lrow[r] = 0.f; }
#pragma unroll
  for (int ni = 0; ni < 4; ni++) of[ni] = (v8f){};

  // staging coordinates
  int kkey = t >> 3, khd = (t & 7) * 8;    // K tile: 32 keys x 64 hd
  int vhd = t >> 2, vko = (t & 3) * 8;     // V^T tile: 64 hd x 32 keys
  const __bf16* Kb = Kg + (size_t)(b * SS + kkey) * DD + h * HD + khd;
  const __bf16* Vb = Vg + ((size_t)(b * HH + h) * HD + vhd) * SS + vko;
  unsigned koff = kkey * KLSTR + khd;
  unsigned voff = vhd * VTSTR + vko;

  const int NKB = SS / 32;
#define STAGEKV(buf, kbi)                                            \
  do {                                                               \
    async_copy_b128(Kb + (size_t)(kbi) * 32 * DD, &Kl[buf][koff]);   \
    async_copy_b128(Vb + (kbi) * 32,              &Vl[buf][voff]);   \
  } while (0)

  STAGEKV(0, 0);
  for (int kb = 0; kb < NKB; kb++) {
    async_wait0();
    __syncthreads();
    int cur = kb & 1;
    if (kb + 1 < NKB) STAGEKV(cur ^ 1, kb + 1);

    // S = Q K^T : two 16x16 C-frags (keys 0..15, 16..31), K-dim 64 in 2 steps
    v8f s[2];
    s[0] = (v8f){}; s[1] = (v8f){};
#pragma unroll
    for (int ni = 0; ni < 2; ni++)
#pragma unroll
      for (int c = 0; c < 2; c++) {
        v16bf bk = load_frag16(&Kl[cur][(ni * 16 + lr) * KLSTR + c * 32 + half * 8], 16);
        s[ni] = __builtin_amdgcn_wmma_f32_16x16x32_bf16(
            false, qf[c], false, bk, (short)0, s[ni], false, false);
      }

    // online softmax (row reductions across 16-lane halves)
    float alpha[8];
#pragma unroll
    for (int r = 0; r < 8; r++) {
      float mx = fmaxf(s[0][r], s[1][r]);
#pragma unroll
      for (int msk = 1; msk < 16; msk <<= 1) mx = fmaxf(mx, __shfl_xor(mx, msk, 32));
      float mnew = fmaxf(mrow[r], mx);
      alpha[r] = __expf(mrow[r] - mnew);
      mrow[r] = mnew;
      float p0 = __expf(s[0][r] - mnew);
      float p1 = __expf(s[1][r] - mnew);
      s[0][r] = p0; s[1][r] = p1;
      float rs = p0 + p1;
#pragma unroll
      for (int msk = 1; msk < 16; msk <<= 1) rs += __shfl_xor(rs, msk, 32);
      lrow[r] = lrow[r] * alpha[r] + rs;
    }
#pragma unroll
    for (int ni = 0; ni < 4; ni++)
#pragma unroll
      for (int r = 0; r < 8; r++) of[ni][r] *= alpha[r];

    // C-layout -> A-layout for P via per-wave LDS patch
    __bf16* pw = Pl[wave];
#pragma unroll
    for (int r = 0; r < 8; r++) {
      pw[(half * 8 + r) * PSTR + lr]      = (__bf16)s[0][r];
      pw[(half * 8 + r) * PSTR + 16 + lr] = (__bf16)s[1][r];
    }
    asm volatile("s_wait_dscnt 0x0" ::: "memory");  // wave-private LDS RAW
    v16bf pa = load_frag16(&pw[lr * PSTR + half * 8], 16);

    // O += P V : 4 WMMAs over hd chunks
#pragma unroll
    for (int ni = 0; ni < 4; ni++) {
      v16bf vb = load_frag16(&Vl[cur][(ni * 16 + lr) * VTSTR + half * 8], 16);
      of[ni] = __builtin_amdgcn_wmma_f32_16x16x32_bf16(
          false, pa, false, vb, (short)0, of[ni], false, false);
    }
  }
#undef STAGEKV

  // normalize by row sums and store bf16
#pragma unroll
  for (int r = 0; r < 8; r++) {
    float inv = 1.f / lrow[r];
    int row = b * SS + q0 + half * 8 + r;
#pragma unroll
    for (int ni = 0; ni < 4; ni++) {
      int col = h * HD + ni * 16 + lr;
      O[(size_t)row * DD + col] = (__bf16)(of[ni][r] * inv);
    }
  }
}

// ---------------------------------------------------------------------------
// Host launcher
// ---------------------------------------------------------------------------
extern "C" void kernel_launch(void* const* d_in, const int* in_sizes, int n_in,
                              void* d_out, int out_size, void* d_ws, size_t ws_size,
                              hipStream_t stream) {
  const float* latents   = (const float*)d_in[0];
  const float* ln1_scale = (const float*)d_in[1];
  const float* wq        = (const float*)d_in[2];
  const float* wk        = (const float*)d_in[3];
  const float* wv        = (const float*)d_in[4];
  const float* q_ns      = (const float*)d_in[5];
  const float* k_ns      = (const float*)d_in[6];
  const float* wo_attn   = (const float*)d_in[7];
  const float* ln2_scale = (const float*)d_in[8];
  const float* wi        = (const float*)d_in[9];
  const float* wo_mlp    = (const float*)d_in[10];
  float* out = (float*)d_out;

  char* ws = (char*)d_ws;
  size_t off = 0;
  auto alloc = [&](size_t bytes) -> void* {
    void* p = ws + off;
    off = (off + bytes + 255) & ~(size_t)255;
    return p;
  };

  const size_t WW = (size_t)DD * DD;       // 589824
  const size_t WI = (size_t)DD * MLPD;     // 2359296
  const size_t XB = (size_t)BS * DD;       // 6291456
  const size_t HBn = (size_t)BS * MLPD;    // 25165824

  __bf16* wq_t = (__bf16*)alloc(WW * 2);   // [N][K] transposed bf16 weights
  __bf16* wk_t = (__bf16*)alloc(WW * 2);
  __bf16* wv_t = (__bf16*)alloc(WW * 2);
  __bf16* wo_t = (__bf16*)alloc(WW * 2);
  __bf16* wi_t = (__bf16*)alloc(WI * 2);
  __bf16* wm_t = (__bf16*)alloc(WI * 2);
  __bf16* xb   = (__bf16*)alloc(XB * 2);
  __bf16* qb   = (__bf16*)alloc(XB * 2);
  __bf16* kb   = (__bf16*)alloc(XB * 2);
  __bf16* vt   = (__bf16*)alloc(XB * 2);   // V transposed [B][H][HD][S]
  __bf16* ob   = (__bf16*)alloc(XB * 2);
  __bf16* yb   = (__bf16*)alloc(XB * 2);
  float*  x2   = (float*) alloc(XB * 4);
  __bf16* hb   = (__bf16*)alloc(HBn * 2);

  // weights: transpose + convert to bf16 (one-time per launch)
  transpose_f32_to_bf16<<<dim3(DD / 32, DD / 32), 256, 0, stream>>>(wq, wq_t, DD, DD);
  transpose_f32_to_bf16<<<dim3(DD / 32, DD / 32), 256, 0, stream>>>(wk, wk_t, DD, DD);
  transpose_f32_to_bf16<<<dim3(DD / 32, DD / 32), 256, 0, stream>>>(wv, wv_t, DD, DD);
  transpose_f32_to_bf16<<<dim3(DD / 32, DD / 32), 256, 0, stream>>>(wo_attn, wo_t, DD, DD);
  transpose_f32_to_bf16<<<dim3(MLPD / 32, DD / 32), 256, 0, stream>>>(wi, wi_t, DD, MLPD);
  transpose_f32_to_bf16<<<dim3(DD / 32, MLPD / 32), 256, 0, stream>>>(wo_mlp, wm_t, MLPD, DD);

  // ln1
  rmsnorm_to_bf16<<<BS, 256, 0, stream>>>(latents, ln1_scale, xb, DD);

  // QKV projections (q,k: bf16 ; v: bf16 transposed for attention)
  dim3 gqkv(DD / GTN, BS / GTM);
  gemm_bf16_wmma<2><<<gqkv, 256, 0, stream>>>(xb, wq_t, BS, DD, DD, nullptr, qb, nullptr);
  gemm_bf16_wmma<2><<<gqkv, 256, 0, stream>>>(xb, wk_t, BS, DD, DD, nullptr, kb, nullptr);
  gemm_bf16_wmma<4><<<gqkv, 256, 0, stream>>>(xb, wv_t, BS, DD, DD, nullptr, vt, nullptr);

  // qk-norm (q also scaled by 1/sqrt(HD))
  int nvec = BS * HH;
  int qkblocks = (nvec * 32 + 255) / 256;
  qk_norm_inplace<<<qkblocks, 256, 0, stream>>>(qb, q_ns, 0.125f, nvec);
  qk_norm_inplace<<<qkblocks, 256, 0, stream>>>(kb, k_ns, 1.0f, nvec);

  // flash attention
  flash_attn_wmma<<<dim3(SS / 128, HH, BB), 256, 0, stream>>>(qb, kb, vt, ob);

  // O-projection + residual -> x2 (f32)
  gemm_bf16_wmma<1><<<gqkv, 256, 0, stream>>>(ob, wo_t, BS, DD, DD, x2, nullptr, latents);

  // ln2
  rmsnorm_to_bf16<<<BS, 256, 0, stream>>>(x2, ln2_scale, yb, DD);

  // MLP up + GELU -> bf16
  dim3 gup(MLPD / GTN, BS / GTM);
  gemm_bf16_wmma<3><<<gup, 256, 0, stream>>>(yb, wi_t, BS, MLPD, DD, nullptr, hb, nullptr);

  // MLP down + residual -> out (f32)
  gemm_bf16_wmma<1><<<gqkv, 256, 0, stream>>>(hb, wm_t, BS, DD, MLPD, out, nullptr, x2);
}
